// RCNLayer_66039417143765
// MI455X (gfx1250) — matrix operand order repaired and checked by reading, not compiled, not run on hardware
//
#include <hip/hip_runtime.h>
#include <hip/hip_fp16.h>

#define NN 100000
#define NE 1600000
#define IN_DIM 128
#define HEADS 4
#define OUT_DIM 32
#define HID (HEADS * OUT_DIM)   // 128
#define EPSF 1e-8f

typedef __attribute__((ext_vector_type(16))) _Float16 v16h;
typedef __attribute__((ext_vector_type(8)))  float    v8f;

// ---- order-preserving float <-> uint encoding for atomicMax on floats ----
__device__ __forceinline__ unsigned f2ord(float f) {
  unsigned u = __float_as_uint(f);
  return (u & 0x80000000u) ? ~u : (u | 0x80000000u);
}
__device__ __forceinline__ float ord2f(unsigned m) {
  unsigned u = (m & 0x80000000u) ? (m & 0x7FFFFFFFu) : ~m;
  return __uint_as_float(u);
}

// ---- 0: init output accumulator + per-node reduction buffers ----
__global__ void init_kernel(float* __restrict__ out_feat,
                            float* __restrict__ denom,
                            unsigned* __restrict__ segmax) {
  const long long stride = (long long)gridDim.x * blockDim.x;
  for (long long i = (long long)blockIdx.x * blockDim.x + threadIdx.x;
       i < (long long)NN * HID; i += stride)
    out_feat[i] = 0.f;
  for (long long i = (long long)blockIdx.x * blockDim.x + threadIdx.x;
       i < (long long)NN * HEADS; i += stride) {
    denom[i]  = 0.f;
    segmax[i] = 0u;   // ordered encoding: smallest possible value
  }
}

// ---- 1: h = x @ W^T via v_wmma_f32_16x16x32_f16 ----
// block = 256 threads = 8 waves; block owns 16 rows; wave w owns cols [16w,16w+16)
__global__ __launch_bounds__(256) void gemm_h_kernel(const float* __restrict__ x,
                                                     const float* __restrict__ W,
                                                     float* __restrict__ h) {
  const int lane    = threadIdx.x & 31;
  const int wave    = threadIdx.x >> 5;
  const int rowBase = blockIdx.x * 16;
  const int colBase = wave * 16;
  const int mn      = lane & 15;                 // A-row / B-col index
  const int koffA   = (lane < 16) ? 0 : 8;       // 16-bit A 16x32 lane layout
  const int koffB   = (lane < 16) ? 0 : 8;       // 16-bit B 32x16 lane layout

  const float* __restrict__ xrow = x + (size_t)(rowBase + mn) * IN_DIM;
  const float* __restrict__ wrow = W + (size_t)(colBase + mn) * IN_DIM;

  v8f acc = {};
#pragma unroll
  for (int kk = 0; kk < IN_DIM; kk += 32) {
    v16h a, b;
#pragma unroll
    for (int i = 0; i < 8; ++i) {
      a[i]     = (_Float16)xrow[kk + koffA + i];
      a[8 + i] = (_Float16)xrow[kk + 16 + koffA + i];
    }
#pragma unroll
    for (int i = 0; i < 8; ++i) {
      b[i]     = (_Float16)wrow[kk + koffB + i];
      b[8 + i] = (_Float16)wrow[kk + 16 + koffB + i];
    }
    acc = __builtin_amdgcn_wmma_f32_16x16x32_f16(false, a, false, b,
                                                 (short)0, acc, false, false);
  }

  // D layout: lane n = lane&15 is col; VGPR v holds row M = v + (lane<16?0:8)
  const int mBase = (lane < 16) ? 0 : 8;
#pragma unroll
  for (int v = 0; v < 8; ++v)
    h[(size_t)(rowBase + mBase + v) * HID + colBase + mn] = acc[v];
}

// ---- 2: per-node attention pre-scores (factorized edge score) ----
__global__ void node_score_kernel(const float* __restrict__ h,
                                  const float* __restrict__ a_src,
                                  const float* __restrict__ a_dst,
                                  float* __restrict__ s_src,
                                  float* __restrict__ s_dst) {
  int t = blockIdx.x * blockDim.x + threadIdx.x;
  if (t >= NN * HEADS) return;
  const int node = t >> 2, head = t & 3;
  const float* hv = h + (size_t)node * HID + head * OUT_DIM;
  const float* as = a_src + head * OUT_DIM;
  const float* ad = a_dst + head * OUT_DIM;
  float ss = 0.f, sd = 0.f;
#pragma unroll
  for (int i = 0; i < OUT_DIM; ++i) {
    float v = hv[i];
    ss += v * as[i];
    sd += v * ad[i];
  }
  s_src[t] = ss;
  s_dst[t] = sd;
}

// ---- 3: edge score (+bias, leaky-relu) staged into alpha; segment max ----
__global__ void edge_score_max_kernel(const int* __restrict__ ei,
                                      const float* __restrict__ w,
                                      const float* __restrict__ s_src,
                                      const float* __restrict__ s_dst,
                                      float* __restrict__ alpha,
                                      unsigned* __restrict__ segmax) {
  int t = blockIdx.x * blockDim.x + threadIdx.x;
  if (t >= NE * HEADS) return;
  const int e = t >> 2, hd = t & 3;
  const int s = ei[e], d = ei[NE + e];
  float sc = s_src[s * HEADS + hd] + s_dst[d * HEADS + hd] + __logf(w[e] + EPSF);
  sc = (sc >= 0.f) ? sc : 0.2f * sc;   // leaky_relu(0.2); TEMPERATURE == 1
  alpha[t] = sc;                       // stage scaled score
  atomicMax(&segmax[d * HEADS + hd], f2ord(sc));
}

// ---- 4: exp(score - segmax) staged into alpha; segment sum ----
__global__ void edge_exp_kernel(const int* __restrict__ ei,
                                const unsigned* __restrict__ segmax,
                                float* __restrict__ alpha,
                                float* __restrict__ denom) {
  int t = blockIdx.x * blockDim.x + threadIdx.x;
  if (t >= NE * HEADS) return;
  const int e = t >> 2, hd = t & 3;
  const int d = ei[NE + e];
  const float m  = ord2f(segmax[d * HEADS + hd]);
  const float ex = __expf(alpha[t] - m);
  alpha[t] = ex;
  atomicAdd(&denom[d * HEADS + hd], ex);
}

// ---- 5: normalize alpha in place (this is the second output) ----
__global__ void edge_norm_kernel(const int* __restrict__ ei,
                                 const float* __restrict__ denom,
                                 float* __restrict__ alpha) {
  int t = blockIdx.x * blockDim.x + threadIdx.x;
  if (t >= NE * HEADS) return;
  const int e = t >> 2, hd = t & 3;
  alpha[t] = alpha[t] / (denom[ei[NE + e] * HEADS + hd] + EPSF);
}

// ---- 6: out[dst] += alpha * h[src]  (L2-resident float atomics) ----
__global__ void aggregate_kernel(const int* __restrict__ ei,
                                 const float* __restrict__ alpha,
                                 const float* __restrict__ h,
                                 float* __restrict__ out_feat) {
  long long t = (long long)blockIdx.x * blockDim.x + threadIdx.x;
  if (t >= (long long)NE * HID) return;
  const int e = (int)(t >> 7), c = (int)(t & 127);
  const int s = ei[e], d = ei[NE + e];
  const float a = alpha[e * HEADS + (c >> 5)];
  atomicAdd(&out_feat[(size_t)d * HID + c], a * h[(size_t)s * HID + c]);
}

extern "C" void kernel_launch(void* const* d_in, const int* in_sizes, int n_in,
                              void* d_out, int out_size, void* d_ws, size_t ws_size,
                              hipStream_t stream) {
  const float* x     = (const float*)d_in[0];
  const float* W     = (const float*)d_in[1];
  const float* a_src = (const float*)d_in[2];
  const float* a_dst = (const float*)d_in[3];
  const int*   ei    = (const int*)d_in[4];
  const float* w     = (const float*)d_in[5];

  float* out_feat = (float*)d_out;                           // [NN, 128]
  float* alpha    = (float*)d_out + (size_t)NN * HID;        // [NE, 4]

  float*    h      = (float*)d_ws;                           // [NN, 128]
  float*    s_src  = h + (size_t)NN * HID;                   // [NN, 4]
  float*    s_dst  = s_src + (size_t)NN * HEADS;             // [NN, 4]
  float*    denom  = s_dst + (size_t)NN * HEADS;             // [NN, 4]
  unsigned* segmax = (unsigned*)(denom + (size_t)NN * HEADS);// [NN, 4]

  init_kernel<<<2048, 256, 0, stream>>>(out_feat, denom, segmax);

  gemm_h_kernel<<<NN / 16, 256, 0, stream>>>(x, W, h);

  node_score_kernel<<<(NN * HEADS + 255) / 256, 256, 0, stream>>>(
      h, a_src, a_dst, s_src, s_dst);

  edge_score_max_kernel<<<(NE * HEADS + 255) / 256, 256, 0, stream>>>(
      ei, w, s_src, s_dst, alpha, segmax);

  edge_exp_kernel<<<(NE * HEADS + 255) / 256, 256, 0, stream>>>(
      ei, segmax, alpha, denom);

  edge_norm_kernel<<<(NE * HEADS + 255) / 256, 256, 0, stream>>>(
      ei, denom, alpha);

  const long long aggN = (long long)NE * HID;
  aggregate_kernel<<<(int)((aggN + 255) / 256), 256, 0, stream>>>(
      ei, alpha, h, out_feat);
}
